// CorrBlock_8177617732070
// MI455X (gfx1250) — matrix operand order repaired and checked by reading, not compiled, not run on hardware
//
#include <hip/hip_runtime.h>
#include <hip/hip_bf16.h>

// ---------------------------------------------------------------------------
// RAFT CorrBlock for MI455X (gfx1250, wave32, WMMA + TDM).
//   Stage 0: fmap (B,C,P) fp32 -> transposed packed bf16 hi/lo (B,P,C).
//            One-time split so the GEMM hot loop has zero conversion VALU.
//   Stage 1: corr[b,p,q] = (1/16) * sum_c f1[c,p] f2[c,q]
//            4096x4096x256 x2 batched GEMM, v_wmma_f32_16x16x32_bf16,
//            3-WMMA bf16 hi/lo split product (~fp32 accuracy).
//            Tiles staged LDS-side by the Tensor Data Mover
//            (tensor_load_to_lds, TENSORcnt, double-buffered 2x40KB LDS).
//   Stage 2: 3x 2x2 average pooling -> 4-level pyramid.
//   Stage 3: 81-point bilinear window gather per level -> (B, N, H, W, 324).
// Workspace: 178 MB pyramid + 16 MB packed bf16 = ~194 MB in d_ws.
// ---------------------------------------------------------------------------

typedef __attribute__((ext_vector_type(16))) __bf16 v16bf;
typedef __attribute__((ext_vector_type(8)))  float  v8f;
typedef unsigned int v4u __attribute__((ext_vector_type(4)));
typedef int          v8i __attribute__((ext_vector_type(8)));
typedef int          v4i __attribute__((ext_vector_type(4)));

#define C_DIM 256
#define P_DIM 4096   // H*W
#define B_DIM 2
#define N_DIM 4
#define H_DIM 64
#define W_DIM 64
#define K_STEP 32
#define LDSK 40      // LDS K-pitch (bf16): 80B rows = 16 data DW + 4 pad DW

__device__ __forceinline__ __bf16 f32_to_bf16_rne(float x) {
    unsigned u = __builtin_bit_cast(unsigned, x);
    unsigned r = u + 0x7FFFu + ((u >> 16) & 1u);
    unsigned short hs = (unsigned short)(r >> 16);
    return __builtin_bit_cast(__bf16, hs);
}
__device__ __forceinline__ float bf16_to_f32(__bf16 h) {
    unsigned short hs = __builtin_bit_cast(unsigned short, h);
    return __builtin_bit_cast(float, ((unsigned)hs) << 16);
}

// ---------------------------------------------------------------------------
// TDM: load a 16-row x 64-byte tile (16 DW rows, global pitch C_DIM*2B) into
// LDS with pitch 80B (pad 4 DW after every 16 DW).  D# per ISA 8.3/8.4.
__device__ __forceinline__ void tdm_load_tile(const __bf16* gsrc, unsigned lds_off) {
    unsigned long long ga = (unsigned long long)(uintptr_t)gsrc;
    v4u g0;
    g0[0] = 1u;                                   // count=1 (valid user D#)
    g0[1] = lds_off;                              // lds_addr (bytes)
    g0[2] = (unsigned)(ga & 0xffffffffu);         // global_addr[31:0]
    g0[3] = (unsigned)((ga >> 32) & 0x1ffffffu)   // global_addr[56:32]
          | (2u << 30);                           // type=2 ("image")
    v8i g1;
    g1[0] = (int)((2u << 16)      // data_size = 4B
                | (1u << 20)      // pad_enable
                | (3u << 22)      // pad_interval: 16 DW between pads
                | (3u << 25));    // pad_amount: 4 DW
    g1[1] = (int)(128u << 16);    // tensor_dim0 = 128 DW/row (lo16 @ [31:16])
    g1[2] = (int)((4096u & 0xffffu) << 16); // dim0 hi=0 | tensor_dim1 lo16
    g1[3] = (int)(16u << 16);     // tensor_dim1 hi=0 | tile_dim0 = 16 DW
    g1[4] = 16;                   // tile_dim1 = 16 rows | tile_dim2 = 0
    g1[5] = 128;                  // tensor_dim0_stride = 128 DW (lo32)
    g1[6] = 0;                    // stride0 hi | stride1 lo
    g1[7] = 0;
    v4i z4 = {0, 0, 0, 0};
#if defined(__clang_major__) && __clang_major__ >= 23
    v8i z8 = {0, 0, 0, 0, 0, 0, 0, 0};
    __builtin_amdgcn_tensor_load_to_lds(g0, g1, z4, z4, z8, 0);
#else
    __builtin_amdgcn_tensor_load_to_lds(g0, g1, z4, z4, 0);
#endif
}

// A fragment (16x32 bf16, MxK): ISA 7.12.2 layout.
__device__ __forceinline__ v16bf load_frag_a(const __bf16* row, int half) {
    v16bf a;
    ((float4*)&a)[0] = *(const float4*)(row + half * 8);
    ((float4*)&a)[1] = *(const float4*)(row + 16 + half * 8);
    return a;
}
// B fragment (32x16 bf16, KxN), LDS holds B^T as [n][k].
__device__ __forceinline__ v16bf load_frag_b(const __bf16* row, int half) {
    v16bf b;
    ((float4*)&b)[0] = *(const float4*)(row + half * 8);
    ((float4*)&b)[1] = *(const float4*)(row + 16 + half * 8);
    return b;
}

// ---------------------------------------------------------------------------
// Stage 0: (B,C,P) fp32 -> (B,P,C) bf16 hi + lo, tiled LDS transpose.
__global__ __launch_bounds__(256)
void convert_transpose(const float* __restrict__ in,
                       __bf16* __restrict__ out_hi,
                       __bf16* __restrict__ out_lo) {
    __shared__ float tile[32][33];
    int b  = blockIdx.z;
    int c0 = blockIdx.y * 32;
    int p0 = blockIdx.x * 32;
    const float* src = in + ((size_t)b * C_DIM + c0) * P_DIM + p0;
#pragma unroll
    for (int i = 0; i < 32; i += 8)
        tile[threadIdx.y + i][threadIdx.x] =
            src[(size_t)(threadIdx.y + i) * P_DIM + threadIdx.x];
    __syncthreads();
    size_t obase = ((size_t)b * P_DIM + p0) * C_DIM + c0;
#pragma unroll
    for (int i = 0; i < 32; i += 8) {
        float v = tile[threadIdx.x][threadIdx.y + i];     // (p=ty+i, c=tx)
        __bf16 h = f32_to_bf16_rne(v);
        size_t o = obase + (size_t)(threadIdx.y + i) * C_DIM + threadIdx.x;
        out_hi[o] = h;
        out_lo[o] = f32_to_bf16_rne(v - bf16_to_f32(h));
    }
}

// ---------------------------------------------------------------------------
// Stage 1: GEMM.  256 threads (8 waves), 128x128 C-tile, K stepped by 32.
// Arrays in smem: 0=A_hi, 1=A_lo, 2=B_hi, 3=B_lo.  Double buffered.
__global__ __launch_bounds__(256)
void corr_gemm_wmma(const __bf16* __restrict__ f1h, const __bf16* __restrict__ f1l,
                    const __bf16* __restrict__ f2h, const __bf16* __restrict__ f2l,
                    float* __restrict__ corr) {
    __shared__ __bf16 smem[2][4][128][LDSK];   // 2 * 4 * 10240B = 80 KB

    const int tid  = threadIdx.x;
    const int lane = tid & 31;
    const int wave = tid >> 5;
    const int half = lane >> 4;
    const int l16  = lane & 15;

    int bx = blockIdx.x;
    const int b   = bx >> 10;
    const int rem = bx & 1023;
    const int p0  = (rem >> 5) << 7;
    const int q0  = (rem & 31) << 7;

    const __bf16* srcs[4] = {
        f1h + ((size_t)b * P_DIM + p0) * C_DIM,
        f1l + ((size_t)b * P_DIM + p0) * C_DIM,
        f2h + ((size_t)b * P_DIM + q0) * C_DIM,
        f2l + ((size_t)b * P_DIM + q0) * C_DIM,
    };
    const int wrow = wave * 16;   // each wave TDM-loads its own 16-row slice

    const int wm = (wave & 3) * 32;   // 4 waves along M
    const int wn = (wave >> 2) * 64;  // 2 waves along N

    v8f acc[2][4];
#pragma unroll
    for (int mi = 0; mi < 2; ++mi)
#pragma unroll
        for (int ni = 0; ni < 4; ++ni)
            acc[mi][ni] = (v8f){0.f, 0.f, 0.f, 0.f, 0.f, 0.f, 0.f, 0.f};

    // Prime buffer 0.
#pragma unroll
    for (int a = 0; a < 4; ++a)
        tdm_load_tile(srcs[a] + (size_t)wrow * C_DIM,
                      (unsigned)(uintptr_t)&smem[0][a][wrow][0]);

    for (int k0 = 0; k0 < C_DIM; k0 += K_STEP) {
        const int buf = (k0 >> 5) & 1;
        if (k0 + K_STEP < C_DIM) {
            // Prefetch next K-slab into the other buffer (previous reads of
            // that buffer were fenced by the barrier at end of last iter).
#pragma unroll
            for (int a = 0; a < 4; ++a)
                tdm_load_tile(srcs[a] + (size_t)wrow * C_DIM + (k0 + K_STEP),
                              (unsigned)(uintptr_t)&smem[buf ^ 1][a][wrow][0]);
            __builtin_amdgcn_s_wait_tensorcnt(4);  // in-order: current 4 done
        } else {
            __builtin_amdgcn_s_wait_tensorcnt(0);
        }
        __syncthreads();   // all waves' slices visible

        v16bf fah[2], fal[2], fbh[4], fbl[4];
#pragma unroll
        for (int mi = 0; mi < 2; ++mi) {
            fah[mi] = load_frag_a(&smem[buf][0][wm + mi * 16 + l16][0], half);
            fal[mi] = load_frag_a(&smem[buf][1][wm + mi * 16 + l16][0], half);
        }
#pragma unroll
        for (int ni = 0; ni < 4; ++ni) {
            fbh[ni] = load_frag_b(&smem[buf][2][wn + ni * 16 + l16][0], half);
            fbl[ni] = load_frag_b(&smem[buf][3][wn + ni * 16 + l16][0], half);
        }

        // 24 WMMAs; product-major order keeps D->C reuse distance = 8.
#pragma unroll
        for (int mi = 0; mi < 2; ++mi)
#pragma unroll
            for (int ni = 0; ni < 4; ++ni)
                acc[mi][ni] = __builtin_amdgcn_wmma_f32_16x16x32_bf16(
                    false, fah[mi], false, fbh[ni], (short)0, acc[mi][ni], false, false);
#pragma unroll
        for (int mi = 0; mi < 2; ++mi)
#pragma unroll
            for (int ni = 0; ni < 4; ++ni)
                acc[mi][ni] = __builtin_amdgcn_wmma_f32_16x16x32_bf16(
                    false, fah[mi], false, fbl[ni], (short)0, acc[mi][ni], false, false);
#pragma unroll
        for (int mi = 0; mi < 2; ++mi)
#pragma unroll
            for (int ni = 0; ni < 4; ++ni)
                acc[mi][ni] = __builtin_amdgcn_wmma_f32_16x16x32_bf16(
                    false, fal[mi], false, fbh[ni], (short)0, acc[mi][ni], false, false);

        __syncthreads();   // fence reads before buf is TDM-overwritten
    }

    const float kScale = 0.0625f;   // 1/sqrt(256)
#pragma unroll
    for (int mi = 0; mi < 2; ++mi)
#pragma unroll
        for (int ni = 0; ni < 4; ++ni) {
            int pg = p0 + wm + mi * 16 + 8 * half;
            int qg = q0 + wn + ni * 16 + l16;
            float* o = corr + ((size_t)b * P_DIM + pg) * P_DIM + qg;
#pragma unroll
            for (int i = 0; i < 8; ++i)
                o[(size_t)i * P_DIM] = acc[mi][ni][i] * kScale;
        }
}

// ---------------------------------------------------------------------------
__global__ void avgpool2x2(const float* __restrict__ in, float* __restrict__ out,
                           int n_maps, int h2, int w2) {
    int idx = blockIdx.x * blockDim.x + threadIdx.x;
    int total = n_maps * h2 * w2;
    if (idx >= total) return;
    int x = idx % w2;
    int y = (idx / w2) % h2;
    int m = idx / (w2 * h2);
    int ww = w2 * 2;
    const float* p = in + (size_t)m * (h2 * 2) * ww;
    float v = p[(2 * y) * ww + 2 * x]     + p[(2 * y) * ww + 2 * x + 1]
            + p[(2 * y + 1) * ww + 2 * x] + p[(2 * y + 1) * ww + 2 * x + 1];
    out[idx] = 0.25f * v;
}

// ---------------------------------------------------------------------------
__global__ void corr_sample(const float* __restrict__ coords,
                            const float* __restrict__ lvl0,
                            const float* __restrict__ lvl1,
                            const float* __restrict__ lvl2,
                            const float* __restrict__ lvl3,
                            float* __restrict__ out, long total) {
    long tid = (long)blockIdx.x * blockDim.x + threadIdx.x;
    if (tid >= total) return;
    int c = (int)(tid % 324);
    long rem = tid / 324;
    int w = (int)(rem % W_DIM); rem /= W_DIM;
    int h = (int)(rem % H_DIM); rem /= H_DIM;
    int n = (int)(rem % N_DIM);
    int b = (int)(rem / N_DIM);
    int l = c / 81;
    int k = c % 81;
    int p = h * W_DIM + w;

    float cx = coords[((((size_t)b * N_DIM + n) * 2 + 0) * H_DIM + h) * W_DIM + w];
    float cy = coords[((((size_t)b * N_DIM + n) * 2 + 1) * H_DIM + h) * W_DIM + w];

    float inv_scale = 1.0f / (float)(1 << l);
    float x = cx * inv_scale + (float)(k / 9 - 4);   // reference quirk: dx=d[k/9]
    float y = cy * inv_scale + (float)(k % 9 - 4);

    int hw = H_DIM >> l;
    const float* cl = (l == 0 ? lvl0 : l == 1 ? lvl1 : l == 2 ? lvl2 : lvl3)
                    + ((size_t)b * P_DIM + p) * hw * hw;

    float x0f = floorf(x), y0f = floorf(y);
    int x0 = (int)x0f, y0 = (int)y0f;
    float wx = x - x0f, wy = y - y0f;

    auto corner = [&](int yi, int xi) -> float {
        if (xi < 0 || xi > hw - 1 || yi < 0 || yi > hw - 1) return 0.f;
        return cl[yi * hw + xi];
    };
    float v = (1.f - wy) * (1.f - wx) * corner(y0, x0)
            + (1.f - wy) * wx         * corner(y0, x0 + 1)
            + wy         * (1.f - wx) * corner(y0 + 1, x0)
            + wy         * wx         * corner(y0 + 1, x0 + 1);
    out[tid] = v;
}

// ---------------------------------------------------------------------------
extern "C" void kernel_launch(void* const* d_in, const int* in_sizes, int n_in,
                              void* d_out, int out_size, void* d_ws, size_t ws_size,
                              hipStream_t stream) {
    const float* fmap1  = (const float*)d_in[0];
    const float* fmap2  = (const float*)d_in[1];
    const float* coords = (const float*)d_in[2];

    float* lvl0 = (float*)d_ws;                                  // (B,P,64,64)
    float* lvl1 = lvl0 + (size_t)B_DIM * P_DIM * 64 * 64;        // (B,P,32,32)
    float* lvl2 = lvl1 + (size_t)B_DIM * P_DIM * 32 * 32;        // (B,P,16,16)
    float* lvl3 = lvl2 + (size_t)B_DIM * P_DIM * 16 * 16;        // (B,P, 8, 8)
    size_t nbf  = (size_t)B_DIM * P_DIM * C_DIM;                 // 2M elems
    __bf16* f1h = (__bf16*)(lvl3 + (size_t)B_DIM * P_DIM * 8 * 8);
    __bf16* f1l = f1h + nbf;
    __bf16* f2h = f1l + nbf;
    __bf16* f2l = f2h + nbf;

    // Stage 0: pack fp32 -> transposed bf16 hi/lo (one-time, ~33MB traffic).
    dim3 tgrid(P_DIM / 32, C_DIM / 32, B_DIM), tblk(32, 8);
    convert_transpose<<<tgrid, tblk, 0, stream>>>(fmap1, f1h, f1l);
    convert_transpose<<<tgrid, tblk, 0, stream>>>(fmap2, f2h, f2l);

    // Stage 1: batched GEMM (2 * 32 * 32 tiles of 128x128).
    corr_gemm_wmma<<<dim3(2048), dim3(256), 0, stream>>>(f1h, f1l, f2h, f2l, lvl0);

    // Stage 2: pyramid.
    int n_maps = B_DIM * P_DIM;
    avgpool2x2<<<(n_maps * 32 * 32 + 255) / 256, 256, 0, stream>>>(lvl0, lvl1, n_maps, 32, 32);
    avgpool2x2<<<(n_maps * 16 * 16 + 255) / 256, 256, 0, stream>>>(lvl1, lvl2, n_maps, 16, 16);
    avgpool2x2<<<(n_maps *  8 *  8 + 255) / 256, 256, 0, stream>>>(lvl2, lvl3, n_maps,  8,  8);

    // Stage 3: bilinear window gather.
    long total = (long)B_DIM * N_DIM * H_DIM * W_DIM * 324;      // 10,616,832
    corr_sample<<<(int)((total + 255) / 256), 256, 0, stream>>>(
        coords, lvl0, lvl1, lvl2, lvl3, (float*)d_out, total);
}
